// GCTN_16509854286211
// MI455X (gfx1250) — matrix operand order repaired and checked by reading, not compiled, not run on hardware
//
#include <hip/hip_runtime.h>
#include <hip/hip_bf16.h>

// Problem constants: B=4, C=64, H=W=128, BLK=64, K=C*9=576
#define HW    16384          // H*W
#define NPIX  65536          // B*H*W
#define KDIM  576            // C*3*3
#define WPERE 36864          // 64 out-ch * 576 (bf16 elems per expert)
#define MTILE 32             // pixels per tile (two 16-row WMMA sub-tiles)

typedef __attribute__((ext_vector_type(16))) __bf16 v16bf;
typedef __attribute__((ext_vector_type(8)))  __bf16 v8bf;
typedef __attribute__((ext_vector_type(8)))  float  v8f;
typedef __attribute__((ext_vector_type(4)))  int    v4i;

typedef __attribute__((address_space(1))) v4i* gptr_v4i;   // global int4*
typedef __attribute__((address_space(3))) v4i* lptr_v4i;   // LDS int4*

#if defined(__has_builtin)
#if __has_builtin(__builtin_amdgcn_global_load_async_to_lds_b128)
#define HAVE_ASYNC_LDS 1
#endif
#endif

// ---------------------------------------------------------------------------
// Kernel 1: convin = x @ Wc^T + bc ; argmax over 64 channels ; histogram
// ---------------------------------------------------------------------------
__global__ __launch_bounds__(256)
void k_argmax(const float* __restrict__ x, const float* __restrict__ Wc,
              const float* __restrict__ bc, int* __restrict__ index,
              unsigned* __restrict__ counts)
{
    __shared__ float sW[64 * 64];
    __shared__ float sb[64];
    const int tid = threadIdx.x;
    for (int i = tid; i < 4096; i += 256) sW[i] = Wc[i];
    if (tid < 64) sb[tid] = bc[tid];
    __syncthreads();

    const int p  = blockIdx.x * 256 + tid;          // pixel id
    const int b  = p >> 14;
    const int hw = p & (HW - 1);
    const float* xb = x + ((size_t)b << 20);        // b*64*16384

    float xv[64];
#pragma unroll
    for (int c = 0; c < 64; ++c) xv[c] = xb[(c << 14) + hw];

    float best = -3.402823466e38f;
    int bi = 0;
    for (int o = 0; o < 64; ++o) {
        float acc = sb[o];
        const float* wr = &sW[o << 6];
#pragma unroll 16
        for (int c = 0; c < 64; ++c) acc = fmaf(xv[c], wr[c], acc);
        if (acc > best) { best = acc; bi = o; }     // strict > == first-max (jnp.argmax)
    }
    index[p] = bi;
    atomicAdd(&counts[bi], 1u);
}

// ---------------------------------------------------------------------------
// Kernel 2: exclusive prefix scan over 64 bucket counts
// ---------------------------------------------------------------------------
__global__ void k_scan(const unsigned* __restrict__ counts,
                       unsigned* __restrict__ offsets,
                       unsigned* __restrict__ cursor)
{
    if (threadIdx.x == 0) {
        unsigned s = 0;
        for (int i = 0; i < 64; ++i) {
            offsets[i] = s; cursor[i] = s; s += counts[i];
        }
    }
}

// ---------------------------------------------------------------------------
// Kernel 3: scatter pixel ids into per-expert buckets
// ---------------------------------------------------------------------------
__global__ __launch_bounds__(256)
void k_scatter(const int* __restrict__ index, unsigned* __restrict__ cursor,
               unsigned* __restrict__ list)
{
    const int p = blockIdx.x * 256 + threadIdx.x;
    const int e = index[p];
    const unsigned pos = atomicAdd(&cursor[e], 1u);
    list[pos] = (unsigned)p;
}

// ---------------------------------------------------------------------------
// Kernel 4: convert Wg (f32, layout [e][o][c][kh][kw]) to bf16, same layout.
// That layout is exactly k-contiguous per output channel: [e][o][k], k=0..575.
// ---------------------------------------------------------------------------
__global__ __launch_bounds__(256)
void k_cvt(const float* __restrict__ Wg, __bf16* __restrict__ wbf, int n4)
{
    int i = blockIdx.x * blockDim.x + threadIdx.x;
    const int stride = gridDim.x * blockDim.x;
    for (; i < n4; i += stride) {
        const float4 v = ((const float4*)Wg)[i];
        __bf16* o = wbf + 4 * (size_t)i;
        o[0] = (__bf16)v.x; o[1] = (__bf16)v.y;
        o[2] = (__bf16)v.z; o[3] = (__bf16)v.w;
    }
}

// ---------------------------------------------------------------------------
// Kernel 5 (flagship): binned im2col conv via v_wmma_f32_16x16x32_bf16.
// grid.x = expert e (64), grid.y = tile-stride workers.
// 128 threads = 4 waves; wave w computes output columns [16w, 16w+16).
// Each tile covers M=32 pixels: two 16-row sub-tiles share one B fragment.
// LDS: sW = 64x576 bf16 expert weights (72KB), sP = 32x576 bf16 patches (36KB)
// ---------------------------------------------------------------------------
__global__ __launch_bounds__(128)
void k_conv(const float* __restrict__ x, const __bf16* __restrict__ wbf,
            const float* __restrict__ bg, const unsigned* __restrict__ offsets,
            const unsigned* __restrict__ counts, const unsigned* __restrict__ list,
            float* __restrict__ out)
{
    extern __shared__ char smem[];
    __bf16* sW = (__bf16*)smem;                       // 64*576 bf16
    __bf16* sP = (__bf16*)(smem + WPERE * 2);         // 32*576 bf16

    const int e   = blockIdx.x;
    const int tid = threadIdx.x;
    const unsigned cnt = counts[e];
    const unsigned off = offsets[e];

    // Stage this expert's weights into LDS.
#if HAVE_ASYNC_LDS
    {
        // GLOBAL_LOAD_ASYNC_TO_LDS_B128: memory -> LDS without VGPR round-trip.
        gptr_v4i src = (gptr_v4i)(wbf + (size_t)e * WPERE);
        lptr_v4i dst = (lptr_v4i)sW;
        for (int i = tid; i < WPERE * 2 / 16; i += 128)
            __builtin_amdgcn_global_load_async_to_lds_b128(src + i, dst + i, 0, 0);
        asm volatile("s_wait_asynccnt 0x0" ::: "memory");
    }
#else
    {
        const uint4* src = (const uint4*)(wbf + (size_t)e * WPERE);
        uint4* dst = (uint4*)sW;
        for (int i = tid; i < WPERE * 2 / 16; i += 128) dst[i] = src[i];
    }
#endif
    __syncthreads();

    const int wv   = tid >> 5;        // wave id: column group
    const int lane = tid & 31;
    const int mrow = lane & 15;       // M row / N column within 16
    const int hi   = lane >> 4;       // lane-half selector

    for (unsigned t = blockIdx.y; t * (unsigned)MTILE < cnt; t += gridDim.y) {
        const unsigned base = off + t * (unsigned)MTILE;
        const int npix = (int)min((unsigned)MTILE, cnt - t * (unsigned)MTILE);
        __builtin_prefetch(list + base, 0, 1);

        // --- im2col gather: 32 pixels x 64 in-channels x 9 taps -> LDS ---
        for (int task = tid; task < MTILE * 64; task += 128) {
            const int p  = task >> 6;
            const int ci = task & 63;
            const unsigned pix = list[base + (unsigned)min(p, npix - 1)];
            const int b = pix >> 14, h = (pix >> 7) & 127, w = pix & 127;
            const float* xb = x + ((((size_t)b << 6) | (size_t)ci) << 14);
            __bf16* dst = sP + p * KDIM + ci * 9;
#pragma unroll
            for (int kh = 0; kh < 3; ++kh) {
                const int hh = h + kh - 1;
                const bool hv = (unsigned)hh < 128u;
#pragma unroll
                for (int kw = 0; kw < 3; ++kw) {
                    const int ww = w + kw - 1;
                    const float v = (hv && (unsigned)ww < 128u) ? xb[(hh << 7) + ww] : 0.f;
                    dst[kh * 3 + kw] = (__bf16)v;
                }
            }
        }
        __syncthreads();

        // --- K=576 WMMA loop: 18 steps x 2 sub-tiles sharing one B frag ---
        v8f acc0 = {};
        v8f acc1 = {};
#pragma unroll
        for (int kk = 0; kk < 18; ++kk) {
            const int kbase = kk * 32;
            // B fragment: column n = 16*wv + mrow, K = kbase + 16*hi + 0..15
            const v16bf bm = *(const v16bf*)(sW + (wv * 16 + mrow) * KDIM + kbase + hi * 16);
            // A fragments: rows mrow and 16+mrow, K = kbase + 16*hi + 0..15
            const v8bf* a0p = (const v8bf*)(sP + mrow * KDIM + kbase + hi * 16);
            const v8bf* a1p = (const v8bf*)(sP + (16 + mrow) * KDIM + kbase + hi * 16);
            const v16bf a0 = __builtin_shufflevector(a0p[0], a0p[1],
                0, 1, 2, 3, 4, 5, 6, 7, 8, 9, 10, 11, 12, 13, 14, 15);
            const v16bf a1 = __builtin_shufflevector(a1p[0], a1p[1],
                0, 1, 2, 3, 4, 5, 6, 7, 8, 9, 10, 11, 12, 13, 14, 15);
            acc0 = __builtin_amdgcn_wmma_f32_16x16x32_bf16(
                false, a0, false, bm, (short)0, acc0, false, false);
            acc1 = __builtin_amdgcn_wmma_f32_16x16x32_bf16(
                false, a1, false, bm, (short)0, acc1, false, false);
        }

        // --- store D tiles + bg gather; lane holds column n, rows r+8*hi ---
        {
            const int n = wv * 16 + mrow;
            const float bgv = bg[(e << 6) + n];
#pragma unroll
            for (int r = 0; r < 8; ++r) {
                const int m0 = r + 8 * hi;
                if (m0 < npix) {
                    const unsigned pix = list[base + (unsigned)m0];
                    const int b = pix >> 14, hw = pix & (HW - 1);
                    out[((((size_t)b << 6) | (size_t)n) << 14) + hw] = acc0[r] + bgv;
                }
                const int m1 = 16 + r + 8 * hi;
                if (m1 < npix) {
                    const unsigned pix = list[base + (unsigned)m1];
                    const int b = pix >> 14, hw = pix & (HW - 1);
                    out[((((size_t)b << 6) | (size_t)n) << 14) + hw] = acc1[r] + bgv;
                }
            }
        }
        __syncthreads();   // protect sP before next tile's gather
    }
}

// ---------------------------------------------------------------------------
// Kernel 6: deterministic per-channel mean / rsqrt(var+eps) over (B,H,W)
// ---------------------------------------------------------------------------
__global__ __launch_bounds__(256)
void k_stats(const float* __restrict__ out, float* __restrict__ mean,
             float* __restrict__ rinv)
{
    __shared__ float ss[256], sq[256];
    const int n = blockIdx.x;
    const int tid = threadIdx.x;
    float s = 0.f, q = 0.f;
    for (int i = tid; i < NPIX; i += 256) {
        const int b = i >> 14, hw = i & (HW - 1);
        const float v = out[((((size_t)b << 6) | (size_t)n) << 14) + hw];
        s += v; q += v * v;
    }
    ss[tid] = s; sq[tid] = q;
    __syncthreads();
    for (int st = 128; st > 0; st >>= 1) {
        if (tid < st) { ss[tid] += ss[tid + st]; sq[tid] += sq[tid + st]; }
        __syncthreads();
    }
    if (tid == 0) {
        const float m = ss[0] * (1.f / NPIX);
        const float var = sq[0] * (1.f / NPIX) - m * m;
        mean[n] = m;
        rinv[n] = rsqrtf(var + 1e-5f);
    }
}

// ---------------------------------------------------------------------------
// Kernel 7: in-place normalize: y = (gc-mean)*rinv*gamma + beta
// ---------------------------------------------------------------------------
__global__ __launch_bounds__(256)
void k_norm(float* __restrict__ out, const float* __restrict__ mean,
            const float* __restrict__ rinv, const float* __restrict__ gamma,
            const float* __restrict__ beta)
{
    const int i = blockIdx.x * 256 + threadIdx.x;   // float4 index
    const int c = (i >> 12) & 63;                   // (i*4 >> 14) & 63
    const float m = mean[c], ri = rinv[c], g = gamma[c], bt = beta[c];
    float4* o4 = (float4*)out;
    float4 v = o4[i];
    v.x = (v.x - m) * ri * g + bt;
    v.y = (v.y - m) * ri * g + bt;
    v.z = (v.z - m) * ri * g + bt;
    v.w = (v.w - m) * ri * g + bt;
    o4[i] = v;
}

// ---------------------------------------------------------------------------
extern "C" void kernel_launch(void* const* d_in, const int* in_sizes, int n_in,
                              void* d_out, int out_size, void* d_ws, size_t ws_size,
                              hipStream_t stream)
{
    const float* x     = (const float*)d_in[0];
    const float* Wc    = (const float*)d_in[1];
    const float* bc    = (const float*)d_in[2];
    const float* Wg    = (const float*)d_in[3];
    const float* bg    = (const float*)d_in[4];
    const float* gamma = (const float*)d_in[5];
    const float* beta  = (const float*)d_in[6];
    float* out = (float*)d_out;

    // Workspace layout (all 256B-aligned)
    uint8_t* w = (uint8_t*)d_ws;
    unsigned* counts  = (unsigned*)(w + 0);
    unsigned* offsets = (unsigned*)(w + 256);
    unsigned* cursor  = (unsigned*)(w + 512);
    float*    meanb   = (float*)   (w + 768);
    float*    rinvb   = (float*)   (w + 1024);
    int*      index   = (int*)     (w + 4096);
    unsigned* list    = (unsigned*)(w + 4096 + NPIX * 4);
    __bf16*   wbf     = (__bf16*)  (w + 4096 + NPIX * 8);   // 4.5MB bf16 weights

    (void)hipMemsetAsync(counts, 0, 256, stream);

    k_argmax <<<NPIX / 256, 256, 0, stream>>>(x, Wc, bc, index, counts);
    k_scan   <<<1, 64, 0, stream>>>(counts, offsets, cursor);
    k_scatter<<<NPIX / 256, 256, 0, stream>>>(index, cursor, list);
    k_cvt    <<<1024, 256, 0, stream>>>(Wg, wbf, (64 * WPERE) / 4);

    const size_t lds = (size_t)(WPERE + MTILE * KDIM) * 2;   // 110592 B
    k_conv   <<<dim3(64, 16), 128, lds, stream>>>(x, wbf, bg, offsets, counts, list, out);

    k_stats  <<<64, 256, 0, stream>>>(out, meanb, rinvb);
    k_norm   <<<NPIX * 64 / 4 / 256, 256, 0, stream>>>(out, meanb, rinvb, gamma, beta);
}
